// Attention_80384607912298
// MI455X (gfx1250) — matrix operand order repaired and checked by reading, not compiled, not run on hardware
//
#include <hip/hip_runtime.h>
#include <hip/hip_bf16.h>
#include <math.h>

// ---------------------------------------------------------------------------
// MI455X (gfx1250) attention block:
//   q = x@wq^T (+RoPE), k = x@wk^T (+RoPE), v = x@wv^T (transposed store),
//   flash attention with GQA (4 q-heads per kv-head), out = y@wo^T.
// All matmuls via v_wmma_f32_16x16x32_bf16 (wave32).
// ---------------------------------------------------------------------------

#define DIM      4096
#define SEQ      2048
#define NHEADS   32
#define NKVHEADS 8
#define HEADDIM  128
#define KVDIM    (NKVHEADS * HEADDIM)   // 1024

typedef __attribute__((ext_vector_type(16))) __bf16 bf16x16;
typedef __attribute__((ext_vector_type(8)))  float  f32x8;

union AB { bf16x16 v; unsigned u[8]; };

__device__ __forceinline__ unsigned short f2bf_bits(float f) {
    union { __bf16 b; unsigned short u; } c;
    c.b = (__bf16)f;
    return c.u;
}

// ------------------------------ f32 -> bf16 --------------------------------
__global__ void f32_to_bf16_kernel(const float* __restrict__ src,
                                   __bf16* __restrict__ dst, long n) {
    long i = (long)blockIdx.x * blockDim.x + threadIdx.x;
    long stride = (long)gridDim.x * blockDim.x;
    for (; i < n; i += stride) dst[i] = (__bf16)src[i];
}

// ------------------------------ WMMA GEMM ----------------------------------
// Out[M,N] = A[M,K] * W[N,K]^T.   Row-major A and W -> both fragment types
// load contiguous bf16 pairs (dword per lane per VGPR), in contiguous k order (same for A and B):
//   A 16x32: lane l (M=l%16), VGPR v -> K = (l/16)*16 + 2*v
//   B 32x16: lane l (N=l%16), VGPR v -> K = (l/16)*16 + 2*v
//   C 16x16: lane l (N=l%16), VGPR r -> M = (l/16)*8 + r
// MODE: 0 = f32 store, 1 = bf16 store, 2 = bf16 store + RoPE, 3 = bf16
// transposed store Out[col*M + row] (builds V^T for the attention kernel).
template <int MODE>
__global__ __launch_bounds__(256, 2)
void gemm_wmma(const __bf16* __restrict__ A, const __bf16* __restrict__ W,
               void* __restrict__ Out, int M, int N, int K,
               const float* __restrict__ fc, const float* __restrict__ fs) {
    const int lane = threadIdx.x & 31;
    const int wave = threadIdx.x >> 5;
    const int ln = lane & 15, lh = lane >> 4;
    const int wm = wave >> 1, wn = wave & 1;          // 4x2 wave grid
    const long m0 = (long)blockIdx.y * 128 + wm * 32; // wave: 32 rows
    const long n0 = (long)blockIdx.x * 128 + wn * 64; // wave: 64 cols

    f32x8 acc[2][4] = {};

    for (int kk = 0; kk < K; kk += 32) {
        AB a[2], b[4];
#pragma unroll
        for (int mi = 0; mi < 2; ++mi) {
            const __bf16* ap = A + (size_t)(m0 + mi * 16 + ln) * K + kk + lh * 16;
#pragma unroll
            for (int v = 0; v < 8; ++v)
                a[mi].u[v] = *reinterpret_cast<const unsigned*>(
                    ap + 2 * v);
        }
#pragma unroll
        for (int ni = 0; ni < 4; ++ni) {
            const __bf16* bp = W + (size_t)(n0 + ni * 16 + ln) * K + kk + lh * 16;
#pragma unroll
            for (int v = 0; v < 8; ++v)
                b[ni].u[v] = *reinterpret_cast<const unsigned*>(bp + 2 * v);
        }
        if (kk + 32 < K) {  // gfx1250 global_prefetch_b8 for next K tile
            __builtin_prefetch(A + (size_t)(m0 + ln) * K + kk + 32, 0, 1);
            __builtin_prefetch(W + (size_t)(n0 + ln) * K + kk + 32, 0, 1);
        }
#pragma unroll
        for (int mi = 0; mi < 2; ++mi)
#pragma unroll
            for (int ni = 0; ni < 4; ++ni)
                acc[mi][ni] = __builtin_amdgcn_wmma_f32_16x16x32_bf16(
                    false, a[mi].v, false, b[ni].v, (short)0, acc[mi][ni],
                    false, false);
    }

#pragma unroll
    for (int mi = 0; mi < 2; ++mi) {
#pragma unroll
        for (int ni = 0; ni < 4; ++ni) {
#pragma unroll
            for (int r = 0; r < 8; ++r) {
                const long row = m0 + mi * 16 + lh * 8 + r;
                const long col = n0 + ni * 16 + ln;
                float val = acc[mi][ni][r];
                if constexpr (MODE == 0) {
                    reinterpret_cast<float*>(Out)[row * N + col] = val;
                } else if constexpr (MODE == 1) {
                    reinterpret_cast<__bf16*>(Out)[row * N + col] = (__bf16)val;
                } else if constexpr (MODE == 2) {
                    // RoPE: pairs (2i,2i+1) are adjacent columns -> adjacent lanes.
                    float partner = __shfl_xor(val, 1, 32);
                    int i = ((int)(col & (HEADDIM - 1))) >> 1;
                    float c = fc[row * (HEADDIM / 2) + i];
                    float s = fs[row * (HEADDIM / 2) + i];
                    float o = (!(col & 1)) ? (val * c - partner * s)   // re' = re*c - im*s
                                           : (partner * s + val * c);  // im' = re*s + im*c
                    reinterpret_cast<__bf16*>(Out)[row * N + col] = (__bf16)o;
                } else {  // MODE 3: transposed bf16 store (V^T)
                    reinterpret_cast<__bf16*>(Out)[col * (long)M + row] = (__bf16)val;
                }
            }
        }
    }
}

// --------------------------- Flash attention -------------------------------
// Grid: (NHEADS, SEQ/128). Block: 256 threads = 8 waves; wave w owns query
// rows [blockIdx.y*128 + w*16, +16). Keys consumed 32 at a time so that the
// P(16x32) x V(32x128) step exactly fits WMMA K=32. P is reshaped via
// wave-private LDS (C-layout -> A-fragment layout).
__global__ __launch_bounds__(256, 2)
void attn_wmma(const __bf16* __restrict__ Q,   // [SEQ, NHEADS*HEADDIM]
               const __bf16* __restrict__ Kb,  // [SEQ, KVDIM]
               const __bf16* __restrict__ Vt,  // [KVDIM, SEQ]  (transposed)
               __bf16* __restrict__ Y) {       // [SEQ, NHEADS*HEADDIM]
    __shared__ unsigned short plds[8][16 * 32];
    const int lane = threadIdx.x & 31;
    const int wave = threadIdx.x >> 5;
    const int ln = lane & 15, lh = lane >> 4;
    const int h = blockIdx.x;
    const int kvh = h >> 2;  // N_REP = 4
    const int q0 = blockIdx.y * 128 + wave * 16;
    const float scale = 0.08838834764831845f;  // 1/sqrt(128)

    // Q fragments for all 4 K-steps of head_dim=128 (kept in registers).
    AB qf[4];
#pragma unroll
    for (int st = 0; st < 4; ++st) {
        const __bf16* qp = Q + (size_t)(q0 + ln) * DIM + h * HEADDIM + st * 32 + lh * 16;
#pragma unroll
        for (int v = 0; v < 8; ++v)
            qf[st].u[v] = *reinterpret_cast<const unsigned*>(
                qp + 2 * v);
    }

    f32x8 o[8] = {};  // 16x128 output accumulator
    float m_i[8], l_i[8];
#pragma unroll
    for (int r = 0; r < 8; ++r) { m_i[r] = -3.0e38f; l_i[r] = 0.f; }

    const int nj = (q0 + 16 + 31) >> 5;  // causal: key blocks of 32
    for (int jb = 0; jb < nj; ++jb) {
        const int j0 = jb * 32;
        // ---- scores S = Q K^T for 32 keys (two 16x16 C tiles) ----
        f32x8 s[2] = {};
#pragma unroll
        for (int t = 0; t < 2; ++t) {
#pragma unroll
            for (int st = 0; st < 4; ++st) {
                AB kb;
                const __bf16* kp = Kb + (size_t)(j0 + t * 16 + ln) * KVDIM +
                                   kvh * HEADDIM + st * 32 + lh * 16;
#pragma unroll
                for (int v = 0; v < 8; ++v)
                    kb.u[v] = *reinterpret_cast<const unsigned*>(kp + 2 * v);
                s[t] = __builtin_amdgcn_wmma_f32_16x16x32_bf16(
                    false, qf[st].v, false, kb.v, (short)0, s[t], false, false);
            }
        }
        // ---- causal mask + online softmax (row stats per lane-half) ----
#pragma unroll
        for (int r = 0; r < 8; ++r) {
            const int row = q0 + lh * 8 + r;
            float v0 = s[0][r] * scale;
            float v1 = s[1][r] * scale;
            if (j0 + ln > row) v0 = -3.0e38f;
            if (j0 + 16 + ln > row) v1 = -3.0e38f;
            float mx = fmaxf(v0, v1);
#pragma unroll
            for (int off = 8; off >= 1; off >>= 1)
                mx = fmaxf(mx, __shfl_xor(mx, off, 32));
            float mnew = fmaxf(m_i[r], mx);
            float sc = __expf(m_i[r] - mnew);
            float p0 = __expf(v0 - mnew);
            float p1 = __expf(v1 - mnew);
            float rs = p0 + p1;
#pragma unroll
            for (int off = 8; off >= 1; off >>= 1)
                rs += __shfl_xor(rs, off, 32);
            l_i[r] = l_i[r] * sc + rs;
            m_i[r] = mnew;
#pragma unroll
            for (int d = 0; d < 8; ++d) o[d][r] *= sc;
            const int m = lh * 8 + r;
            plds[wave][m * 32 + ln] = f2bf_bits(p0);
            plds[wave][m * 32 + 16 + ln] = f2bf_bits(p1);
        }
        // ---- reload P in A-fragment layout from wave-private LDS ----
        AB pf;
#pragma unroll
        for (int v = 0; v < 8; ++v) {
            int k = lh * 16 + 2 * v;
            pf.u[v] = *reinterpret_cast<const unsigned*>(&plds[wave][ln * 32 + k]);
        }
        // ---- O += P * V (V^T layout makes B pairs contiguous over keys) ----
#pragma unroll
        for (int d = 0; d < 8; ++d) {
            AB vb;
            const __bf16* vp = Vt + (size_t)(kvh * HEADDIM + d * 16 + ln) * SEQ +
                               j0 + lh * 16;
#pragma unroll
            for (int v = 0; v < 8; ++v)
                vb.u[v] = *reinterpret_cast<const unsigned*>(vp + 2 * v);
            o[d] = __builtin_amdgcn_wmma_f32_16x16x32_bf16(
                false, pf.v, false, vb.v, (short)0, o[d], false, false);
        }
    }
    // ---- normalize and store y (bf16 for the output projection) ----
#pragma unroll
    for (int r = 0; r < 8; ++r) {
        float inv = 1.0f / l_i[r];
        const long row = q0 + lh * 8 + r;
#pragma unroll
        for (int d = 0; d < 8; ++d)
            Y[row * DIM + h * HEADDIM + d * 16 + ln] = (__bf16)(o[d][r] * inv);
    }
}

// ------------------------------- launcher ----------------------------------
extern "C" void kernel_launch(void* const* d_in, const int* in_sizes, int n_in,
                              void* d_out, int out_size, void* d_ws, size_t ws_size,
                              hipStream_t stream) {
    (void)in_sizes; (void)n_in; (void)out_size; (void)ws_size;
    const float* x  = (const float*)d_in[0];
    const float* wq = (const float*)d_in[1];
    const float* wk = (const float*)d_in[2];
    const float* wv = (const float*)d_in[3];
    const float* wo = (const float*)d_in[4];
    const float* fc = (const float*)d_in[5];
    const float* fs = (const float*)d_in[6];

    // Workspace carve-up (bf16 buffers), ~136 MiB total.
    char* ws = (char*)d_ws;
    size_t off = 0;
    __bf16* x_bf  = (__bf16*)(ws + off); off += (size_t)SEQ * DIM * 2;     // 16 MiB
    __bf16* wq_bf = (__bf16*)(ws + off); off += (size_t)DIM * DIM * 2;     // 32 MiB
    __bf16* wk_bf = (__bf16*)(ws + off); off += (size_t)KVDIM * DIM * 2;   //  8 MiB
    __bf16* wv_bf = (__bf16*)(ws + off); off += (size_t)KVDIM * DIM * 2;   //  8 MiB
    __bf16* wo_bf = (__bf16*)(ws + off); off += (size_t)DIM * DIM * 2;     // 32 MiB
    __bf16* q_bf  = (__bf16*)(ws + off); off += (size_t)SEQ * DIM * 2;     // 16 MiB
    __bf16* k_bf  = (__bf16*)(ws + off); off += (size_t)SEQ * KVDIM * 2;   //  4 MiB
    __bf16* vt_bf = (__bf16*)(ws + off); off += (size_t)KVDIM * SEQ * 2;   //  4 MiB
    __bf16* y_bf  = (__bf16*)(ws + off); off += (size_t)SEQ * DIM * 2;     // 16 MiB

    // 1) downcast inputs/weights to bf16
    f32_to_bf16_kernel<<<4096, 256, 0, stream>>>(x,  x_bf,  (long)SEQ * DIM);
    f32_to_bf16_kernel<<<4096, 256, 0, stream>>>(wq, wq_bf, (long)DIM * DIM);
    f32_to_bf16_kernel<<<4096, 256, 0, stream>>>(wk, wk_bf, (long)KVDIM * DIM);
    f32_to_bf16_kernel<<<4096, 256, 0, stream>>>(wv, wv_bf, (long)KVDIM * DIM);
    f32_to_bf16_kernel<<<4096, 256, 0, stream>>>(wo, wo_bf, (long)DIM * DIM);

    dim3 blk(256);
    // 2) projections (RoPE fused into Q/K epilogue; V stored transposed)
    gemm_wmma<2><<<dim3(DIM / 128, SEQ / 128), blk, 0, stream>>>(
        x_bf, wq_bf, q_bf, SEQ, DIM, DIM, fc, fs);
    gemm_wmma<2><<<dim3(KVDIM / 128, SEQ / 128), blk, 0, stream>>>(
        x_bf, wk_bf, k_bf, SEQ, KVDIM, DIM, fc, fs);
    gemm_wmma<3><<<dim3(KVDIM / 128, SEQ / 128), blk, 0, stream>>>(
        x_bf, wv_bf, vt_bf, SEQ, KVDIM, DIM, nullptr, nullptr);
    // 3) causal GQA flash attention
    attn_wmma<<<dim3(NHEADS, SEQ / 128), blk, 0, stream>>>(q_bf, k_bf, vt_bf, y_bf);
    // 4) output projection -> f32 d_out
    gemm_wmma<0><<<dim3(DIM / 128, SEQ / 128), blk, 0, stream>>>(
        y_bf, wo_bf, d_out, SEQ, DIM, DIM, nullptr, nullptr);
}
